// CrossUFormer_2774548873757
// MI455X (gfx1250) — matrix operand order repaired and checked by reading, not compiled, not run on hardware
//
#include <hip/hip_runtime.h>
#include <hip/hip_bf16.h>

// ---------------------------------------------------------------------------
// CrossUFormer block for MI455X (gfx1250, wave32, WMMA).
// Pipeline: bias-MLP table -> fused QKV projection (WMMA) ->
//           windowed attention (WMMA QK^T + WMMA PV, fused bias/mask/softmax,
//           roll folded into indexing) -> fused depthwise5x5 + residual +
//           output projection (WMMA).
// All matrix math uses v_wmma_f32_16x16x32_f16 (f16 in, f32 accumulate).
// All WMMA operand tiles are staged in LDS in fragment-native order so every
// fragment load is contiguous ds_load_b128 traffic (no half-word repacking).
// ---------------------------------------------------------------------------

typedef __attribute__((ext_vector_type(16))) _Float16 v16h;
typedef __attribute__((ext_vector_type(8)))  float    v8f;

#define HRES 256
#define WRES 256
#define HW   (HRES * WRES)        // 65536
#define BATCH 4
#define DIM  96
#define HEADS 3
#define HD   32
#define WSZ  8
#define SSH  4
#define NWIN 64                   // 8*8 elements per window
#define QKVC (3 * DIM)            // 288
#define PIXTILE 128               // pixels per block in GEMM kernels
#define SCALE_Q 0.17677669529663687f   // 32^-0.5

// ---------------- WMMA fragment helpers (ISA 7.12.2 layouts) ----------------

// A (16x32, f16): lane<16 -> row M=l, K in {k0..k0+7, k0+16..k0+23}
//                 lane>=16 -> row M=l-16, K in {k0+8..k0+15, k0+24..k0+31}
__device__ __forceinline__ v16h load_frag_a(const _Float16* m, int ld,
                                            int row0, int k0, int lane) {
    const int l = lane & 15, hi = lane >> 4;
    const _Float16* r = m + (row0 + l) * ld;
    v16h a;
#pragma unroll
    for (int i = 0; i < 8; ++i) a[i]     = r[k0 + hi * 8 + i];
#pragma unroll
    for (int i = 0; i < 8; ++i) a[8 + i] = r[k0 + 16 + hi * 8 + i];
    return a;
}

// B (32x16, f16) from an [N][K] row-major buffer:
// lane holds column N = n0+(lane&15), K in {k0+hi*8..+7, k0+16+hi*8..+7}
// -> two aligned 16-byte chunks per lane (two ds_load_b128).
__device__ __forceinline__ v16h load_frag_b_nk(const _Float16* m, int ld,
                                               int n0, int k0, int lane) {
    const int n  = n0 + (lane & 15);
    const _Float16* r = m + n * ld + k0 + (lane >> 4) * 8;
    v16h b;
#pragma unroll
    for (int i = 0; i < 8; ++i) b[i] = r[i];
#pragma unroll
    for (int i = 0; i < 8; ++i) b[8 + i] = r[16 + i];
    return b;
}

__device__ __forceinline__ v8f wmma_f16(v16h a, v16h b, v8f c) {
    // (neg_a, A, neg_b, B, c_mod, C, reuse_a, reuse_b)
    return __builtin_amdgcn_wmma_f32_16x16x32_f16(false, a, false, b,
                                                  (short)0, c, false, false);
}

// ---------------- small helpers ----------------

__device__ __forceinline__ float sgnlog(float v) {
    float a = __builtin_fabsf(v);
    float s = (v > 0.f) ? 1.f : ((v < 0.f) ? -1.f : 0.f);
    return s * log1pf(a);
}

__device__ __forceinline__ int rid(int v) {
    return v < (HRES - WSZ) ? 0 : (v < (HRES - SSH) ? 1 : 2);
}
__device__ __forceinline__ int region(int wh, int ww, int e) {
    return 3 * rid(wh * WSZ + (e >> 3)) + rid(ww * WSZ + (e & 7));
}
__device__ __forceinline__ int refl(int p) {
    return p < 0 ? -p : (p > HRES - 1 ? 2 * (HRES - 1) - p : p);
}

// ---------------- kernel 0: relative-position bias table [3][64][64] --------

__global__ __launch_bounds__(256) void k_bias_mlp(
    const float* __restrict__ Wm1, const float* __restrict__ bm1,
    const float* __restrict__ Wm2, const float* __restrict__ bm2,
    float* __restrict__ bias) {
    int p = blockIdx.x * blockDim.x + threadIdx.x;
    if (p >= NWIN * NWIN) return;
    int n = p >> 6, m = p & 63;
    float r0 = sgnlog((float)((n >> 3) - (m >> 3)));
    float r1 = sgnlog((float)((n & 7) - (m & 7)));
    float a0 = bm2[0], a1 = bm2[1], a2 = bm2[2];
    for (int u = 0; u < 256; ++u) {
        float h = r0 * Wm1[u] + r1 * Wm1[256 + u] + bm1[u];
        h = h > 0.f ? h : 0.f;
        a0 += h * Wm2[u * 3 + 0];
        a1 += h * Wm2[u * 3 + 1];
        a2 += h * Wm2[u * 3 + 2];
    }
    bias[0 * 4096 + p] = a0;
    bias[1 * 4096 + p] = a1;
    bias[2 * 4096 + p] = a2;
}

// ---------------- kernel 1: fused QKV projection (WMMA GEMM) ----------------
// out[pix][oc] = W_cat[oc][:] . x[pix][:] + b_cat[oc], oc in [0,288)
// qkv stored f16 channels-last; V additionally stored f32 NCHW for dwconv.

__global__ __launch_bounds__(256) void k_qkv_proj(
    const float* __restrict__ x,
    const float* __restrict__ Wqk, const float* __restrict__ bqk,
    const float* __restrict__ Wv,  const float* __restrict__ bv,
    _Float16* __restrict__ qkv, float* __restrict__ Vf) {
    __shared__ _Float16 sW[QKVC * DIM];     // 288x96 f16 (A: oc-major) = 55 KB
    __shared__ _Float16 sX[PIXTILE * DIM];  // 128x96 f16 (B: pix-major) = 24 KB
    __shared__ float    sB[QKVC];

    const int tid = threadIdx.x;
    const int p0  = blockIdx.x * PIXTILE;
    const int b   = p0 >> 16;               // 128 | HW -> batch uniform
    const int hw0 = p0 & (HW - 1);

    for (int t = tid; t < QKVC * DIM; t += 256)
        sW[t] = (_Float16)(t < 2 * DIM * DIM ? Wqk[t] : Wv[t - 2 * DIM * DIM]);
    for (int t = tid; t < QKVC; t += 256)
        sB[t] = t < 2 * DIM ? bqk[t] : bv[t - 2 * DIM];
    for (int t = tid; t < DIM * PIXTILE; t += 256) {
        int c = t >> 7, i = t & (PIXTILE - 1);        // coalesced global read
        sX[i * DIM + c] =                              // fragment-native write
            (_Float16)x[((size_t)b * DIM + c) * HW + hw0 + i];
    }
    // CDNA5 speculative prefetch of the next tile's first channel row.
    if (blockIdx.x + 1 < gridDim.x)
        __builtin_prefetch(&x[(size_t)((p0 + PIXTILE) >> 16) * DIM * HW +
                              ((p0 + PIXTILE) & (HW - 1))], 0, 1);
    __syncthreads();

    const int wave = tid >> 5, lane = tid & 31;
    const int l = lane & 15, hi = lane >> 4;
    const int pt = wave;                    // 8 waves x 16-pixel tiles = 128

    for (int ot = 0; ot < QKVC / 16; ++ot) {
        v8f acc = {};
#pragma unroll
        for (int kk = 0; kk < DIM; kk += 32) {
            v16h a  = load_frag_a(sW, DIM, ot * 16, kk, lane);
            v16h bf = load_frag_b_nk(sX, DIM, pt * 16, kk, lane);
            acc = wmma_f16(a, bf, acc);
        }
        const int pix_i = pt * 16 + l;
        const size_t pix = (size_t)p0 + pix_i;
        const int hw = hw0 + pix_i;
#pragma unroll
        for (int r = 0; r < 8; ++r) {
            int oc = ot * 16 + hi * 8 + r;
            float v = acc[r] + sB[oc];
            qkv[pix * QKVC + oc] = (_Float16)v;   // 8 contiguous halves/lane
            if (oc >= 2 * DIM)
                Vf[((size_t)b * DIM + (oc - 2 * DIM)) * HW + hw] = v;
        }
    }
}

// ---------------- kernel 2: windowed attention (one block per window*head) --

__global__ __launch_bounds__(128) void k_attn(
    const _Float16* __restrict__ qkv, const float* __restrict__ bias,
    float* __restrict__ attn_out) {
    __shared__ _Float16 sQ[NWIN * HD];       // [n][d]  (A for QK^T)
    __shared__ _Float16 sK[NWIN * HD];       // [m][d]  (B^T for QK^T)
    __shared__ _Float16 sVt[HD * NWIN];      // [d][m]  (B^T for PV)
    __shared__ float    sP[NWIN * NWIN];     // logits f32
    __shared__ _Float16 sPh[NWIN * NWIN];    // probs f16 (A for PV)

    const int bx   = blockIdx.x;
    const int head = bx % HEADS;
    const int win  = bx / HEADS;
    const int b    = win >> 10;              // 1024 windows per batch
    const int wi   = win & 1023;
    const int wh   = wi >> 5, ww = wi & 31;
    const int tid  = threadIdx.x;

    // Load q (pre-scaled), k, v with the (-4,-4) roll folded into indexing.
    for (int t = tid; t < NWIN * HD; t += 128) {
        int n = t >> 5, d = t & 31;
        int hs = (wh * WSZ + (n >> 3) + SSH) & (HRES - 1);
        int ws = (ww * WSZ + (n & 7)  + SSH) & (WRES - 1);
        size_t base = ((size_t)b * HW + hs * WRES + ws) * QKVC;
        sQ[t] = (_Float16)((float)qkv[base + head * HD + d] * SCALE_Q);
        sK[t] = qkv[base + DIM + head * HD + d];
        sVt[d * NWIN + n] = qkv[base + 2 * DIM + head * HD + d];  // transpose
    }
    __syncthreads();

    const int wave = tid >> 5, lane = tid & 31;
    const int l = lane & 15, hi = lane >> 4;
    const int n0 = wave * 16;                // 4 waves cover 64 rows

    // logits = q k^T  (K=32 -> one WMMA per 16x16 tile), fuse bias + mask
    {
        v16h a = load_frag_a(sQ, HD, n0, 0, lane);
#pragma unroll
        for (int mt = 0; mt < 4; ++mt) {
            v16h bf = load_frag_b_nk(sK, HD, mt * 16, 0, lane);
            v8f c = {};
            c = wmma_f16(a, bf, c);
            const int m   = mt * 16 + l;
            const int idm = region(wh, ww, m);
#pragma unroll
            for (int r = 0; r < 8; ++r) {
                int n = n0 + hi * 8 + r;
                float msk = (region(wh, ww, n) != idm) ? -100.0f : 0.0f;
                sP[n * NWIN + m] =
                    c[r] + bias[head * 4096 + n * NWIN + m] + msk;
            }
        }
    }
    __syncthreads();

    // row softmax -> f16 probabilities
    if (tid < NWIN) {
        float mx = -1e30f;
        for (int m = 0; m < NWIN; ++m) mx = fmaxf(mx, sP[tid * NWIN + m]);
        float sum = 0.f;
        for (int m = 0; m < NWIN; ++m) sum += __expf(sP[tid * NWIN + m] - mx);
        float inv = 1.f / sum;
        for (int m = 0; m < NWIN; ++m)
            sPh[tid * NWIN + m] =
                (_Float16)(__expf(sP[tid * NWIN + m] - mx) * inv);
    }
    __syncthreads();

    // out = P v   (M=64 rows split over waves, N=32, K=64 -> 2 WMMAs/tile)
    v16h a0 = load_frag_a(sPh, NWIN, n0, 0,  lane);
    v16h a1 = load_frag_a(sPh, NWIN, n0, 32, lane);
#pragma unroll
    for (int dt = 0; dt < 2; ++dt) {
        v8f c = {};
        c = wmma_f16(a0, load_frag_b_nk(sVt, NWIN, dt * 16, 0,  lane), c);
        c = wmma_f16(a1, load_frag_b_nk(sVt, NWIN, dt * 16, 32, lane), c);
        const int ch = head * HD + dt * 16 + l;
#pragma unroll
        for (int r = 0; r < 8; ++r) {
            int n  = n0 + hi * 8 + r;
            int ho = (wh * WSZ + (n >> 3) + SSH) & (HRES - 1);  // un-roll +4
            int wo = (ww * WSZ + (n & 7)  + SSH) & (WRES - 1);
            attn_out[((size_t)b * HW + ho * WRES + wo) * DIM + ch] = c[r];
        }
    }
}

// ---------------- kernel 3: dwconv5x5(reflect) + residual + proj (WMMA) -----

__global__ __launch_bounds__(256) void k_final(
    const float* __restrict__ Vf, const float* __restrict__ attn_out,
    const float* __restrict__ Wdw, const float* __restrict__ bdw,
    const float* __restrict__ Wp,  const float* __restrict__ bp,
    float* __restrict__ out) {
    __shared__ _Float16 sWp[DIM * DIM];      // 96x96 f16 (A) = 18 KB
    __shared__ _Float16 sS[PIXTILE * DIM];   // summed features, pix-major (B)
    __shared__ float    sBp[DIM];

    const int tid = threadIdx.x;
    const int p0  = blockIdx.x * PIXTILE;
    const int b   = p0 >> 16;
    const int hw0 = p0 & (HW - 1);

    for (int t = tid; t < DIM * DIM; t += 256) sWp[t] = (_Float16)Wp[t];
    if (tid < DIM) sBp[tid] = bp[tid];

    for (int t = tid; t < DIM * PIXTILE; t += 256) {
        int c = t >> 7, i = t & (PIXTILE - 1);
        int hw = hw0 + i, h = hw >> 8, w = hw & (WRES - 1);
        const float* vc = Vf + ((size_t)b * DIM + c) * HW;
        const float* wk = Wdw + c * 25;
        float acc = bdw[c];
#pragma unroll
        for (int dy = 0; dy < 5; ++dy) {
            int hh = refl(h + dy - 2);
#pragma unroll
            for (int dx = 0; dx < 5; ++dx)
                acc += vc[hh * WRES + refl(w + dx - 2)] * wk[dy * 5 + dx];
        }
        acc += attn_out[((size_t)p0 + i) * DIM + c];
        sS[i * DIM + c] = (_Float16)acc;               // fragment-native write
    }
    __syncthreads();

    const int wave = tid >> 5, lane = tid & 31;
    const int l = lane & 15, hi = lane >> 4;
    const int pt = wave;

    for (int ot = 0; ot < DIM / 16; ++ot) {
        v8f acc = {};
#pragma unroll
        for (int kk = 0; kk < DIM; kk += 32) {
            v16h a  = load_frag_a(sWp, DIM, ot * 16, kk, lane);
            v16h bf = load_frag_b_nk(sS, DIM, pt * 16, kk, lane);
            acc = wmma_f16(a, bf, acc);
        }
        const int hw = hw0 + pt * 16 + l;
#pragma unroll
        for (int r = 0; r < 8; ++r) {
            int oc = ot * 16 + hi * 8 + r;
            out[((size_t)b * DIM + oc) * HW + hw] = acc[r] + sBp[oc];
        }
    }
}

// ---------------- launcher ----------------

extern "C" void kernel_launch(void* const* d_in, const int* in_sizes, int n_in,
                              void* d_out, int out_size, void* d_ws,
                              size_t ws_size, hipStream_t stream) {
    (void)in_sizes; (void)n_in; (void)out_size; (void)ws_size;

    const float* x   = (const float*)d_in[0];
    const float* Wv  = (const float*)d_in[1];
    const float* bv  = (const float*)d_in[2];
    const float* Wqk = (const float*)d_in[3];
    const float* bqk = (const float*)d_in[4];
    const float* Wm1 = (const float*)d_in[5];
    const float* bm1 = (const float*)d_in[6];
    const float* Wm2 = (const float*)d_in[7];
    const float* bm2 = (const float*)d_in[8];
    const float* Wdw = (const float*)d_in[9];
    const float* bdw = (const float*)d_in[10];
    const float* Wp  = (const float*)d_in[11];
    const float* bp  = (const float*)d_in[12];
    float* out = (float*)d_out;

    // Workspace layout (all 256B-aligned by construction):
    const size_t QKV_BYTES = (size_t)BATCH * HW * QKVC * sizeof(_Float16); // 151 MB
    const size_t V_BYTES   = (size_t)BATCH * DIM * HW * sizeof(float);     // 100 MB
    const size_t AO_BYTES  = (size_t)BATCH * HW * DIM * sizeof(float);     // 100 MB
    char* w = (char*)d_ws;
    _Float16* qkv      = (_Float16*)w;
    float*    Vf       = (float*)(w + QKV_BYTES);
    float*    attn_out = (float*)(w + QKV_BYTES + V_BYTES);
    float*    bias     = (float*)(w + QKV_BYTES + V_BYTES + AO_BYTES);

    const int pix_blocks = (BATCH * HW) / PIXTILE;        // 2048

    k_bias_mlp<<<(NWIN * NWIN + 255) / 256, 256, 0, stream>>>(Wm1, bm1, Wm2,
                                                              bm2, bias);
    k_qkv_proj<<<pix_blocks, 256, 0, stream>>>(x, Wqk, bqk, Wv, bv, qkv, Vf);
    k_attn<<<BATCH * 1024 * HEADS, 128, 0, stream>>>(qkv, bias, attn_out);
    k_final<<<pix_blocks, 256, 0, stream>>>(Vf, attn_out, Wdw, bdw, Wp, bp,
                                            out);
}
